// LocalClipLoss_15410342658098
// MI455X (gfx1250) — matrix operand order repaired and checked by reading, not compiled, not run on hardware
//
#include <hip/hip_runtime.h>
#include <hip/hip_bf16.h>

typedef __attribute__((ext_vector_type(16))) __bf16 v16bf;
typedef __attribute__((ext_vector_type(8)))  __bf16 v8bf;
typedef __attribute__((ext_vector_type(8)))  float  v8f;

#define NB    32      // batch (= #captions)
#define SDIM  196     // image regions
#define CDIM  768     // feature dim
#define WDIM  97      // max caption length
#define SP2   208     // attn2t stored S columns (13 tiles)
#define WP    112     // padded W (7 tiles of 16)
#define NT_W  7
#define NWAVES 14
#define THREADS 448

// ---------------------------------------------------------------------------
// Fused per-(n,b) kernel: scores GEMM (bf16x3 WMMA), softmax over q,
// softmax over s, wei GEMM (bf16 WMMA), cosine sim, exp-sum -> simlog[n,b].
// Also writes att_maps rows when n == b.
// ---------------------------------------------------------------------------
__global__ __launch_bounds__(THREADS) void damsm_attn_kernel(
    const float* __restrict__ img,     // [B,S,C]
    const float* __restrict__ words,   // [N,C,W]
    const int*   __restrict__ cap_lens,
    const float* __restrict__ t1p, const float* __restrict__ t2p,
    float* __restrict__ out,           // [0]=loss (written by kernel2), 1.. att_maps [N,W,S]
    float* __restrict__ simlog)        // [N*B]
{
  const int b   = blockIdx.x;
  const int n   = blockIdx.y;
  const int tid = threadIdx.x;
  const int wid = tid >> 5;
  const int lane = tid & 31;
  const int l15 = lane & 15;
  const int hl  = lane >> 4;           // lane half (0/1)
  const float temp1 = t1p[0];
  const float temp2 = t2p[0];
  const int capn = cap_lens[n];

  __shared__ float s_red[WP * 3];
  __shared__ float s_simsum;
  __shared__ __attribute__((aligned(32))) __bf16 s_attn2t[WP * SP2];    // 46592 B
  __shared__ __attribute__((aligned(32))) unsigned char s_u[14336];      // staging/reduce union

  for (int i = tid; i < WP * 3; i += THREADS) s_red[i] = 0.f;
  if (tid == 0) s_simsum = 0.f;

  // =========================================================================
  // Phase 1: scores[s,q] = img[b] x words[n], K = 768 (24 steps of 32).
  // Wave w owns s-tile rows [16w, 16w+16). 7 q-tiles accumulated in regs.
  // bf16x3 split for fp32-class accuracy: AhiBhi + AhiBlo + AloBhi.
  // =========================================================================
  v8f acc[NT_W];
  #pragma unroll
  for (int t = 0; t < NT_W; ++t) acc[t] = {};

  __bf16* swh = (__bf16*)s_u;              // words frag hi  [7][32][16]
  __bf16* swl = (__bf16*)(s_u + 7168);     // words frag lo

  const int s0   = wid * 16;
  const int srow = s0 + l15;
  const bool svalidA = (srow < SDIM);
  const size_t imgRow = ((size_t)b * SDIM + (svalidA ? srow : 0)) * CDIM;

  for (int kt = 0; kt < 24; ++kt) {
    const int k0 = kt * 32;
    // ---- cooperative stage of words k-slice into B-fragment layout ----
    #pragma unroll
    for (int i = 0; i < 8; ++i) {
      int e = tid + i * THREADS;                 // 0..3583
      int qt = e >> 9, rem = e & 511, ln = rem >> 4, j = rem & 15;
      int q = qt * 16 + (ln & 15);
      int k = k0 + ((ln >> 4) << 3) + ((j < 8) ? j : (8 + j));
      float v = (q < WDIM) ? words[((size_t)n * CDIM + k) * WDIM + q] : 0.f;
      __bf16 h = (__bf16)v;
      swh[e] = h;
      swl[e] = (__bf16)(v - (float)h);
    }
    __syncthreads();

    // ---- A fragment straight from global (two contiguous 8-float runs) ----
    const int base = k0 + 8 * hl;
    v16bf ahi, alo;
    #pragma unroll
    for (int j = 0; j < 16; ++j) {
      int kk = base + ((j < 8) ? j : (8 + j));   // K = base+[0..7], base+16+[0..7]
      float x = svalidA ? img[imgRow + kk] : 0.f;
      __bf16 h = (__bf16)x;
      ahi[j] = h;
      alo[j] = (__bf16)(x - (float)h);
    }

    #pragma unroll
    for (int t = 0; t < NT_W; ++t) {
      v16bf bh = *(const v16bf*)(swh + (t << 9) + (lane << 4));
      v16bf bl = *(const v16bf*)(swl + (t << 9) + (lane << 4));
      acc[t] = __builtin_amdgcn_wmma_f32_16x16x32_bf16(false, ahi, false, bh, (short)0, acc[t], false, false);
      acc[t] = __builtin_amdgcn_wmma_f32_16x16x32_bf16(false, ahi, false, bl, (short)0, acc[t], false, false);
      acc[t] = __builtin_amdgcn_wmma_f32_16x16x32_bf16(false, alo, false, bh, (short)0, acc[t], false, false);
    }
    __syncthreads();
  }

  // =========================================================================
  // Softmax over q (per s-row, masked by cap_len). Row = fixed (r, half);
  // values spread over 16 lanes x 7 tiles -> shfl width-16 reductions.
  // =========================================================================
  #pragma unroll
  for (int r = 0; r < 8; ++r) {
    float m = -INFINITY;
    #pragma unroll
    for (int t = 0; t < NT_W; ++t) {
      int q = t * 16 + l15;
      if (q < capn) m = fmaxf(m, acc[t][r]);
    }
    #pragma unroll
    for (int off = 1; off < 16; off <<= 1) m = fmaxf(m, __shfl_xor(m, off, 16));
    float s = 0.f;
    #pragma unroll
    for (int t = 0; t < NT_W; ++t) {
      int q = t * 16 + l15;
      float e = (q < capn) ? __expf(acc[t][r] - m) : 0.f;
      acc[t][r] = e;
      s += e;
    }
    #pragma unroll
    for (int off = 1; off < 16; off <<= 1) s += __shfl_xor(s, off, 16);
    float inv = 1.f / s;
    #pragma unroll
    for (int t = 0; t < NT_W; ++t) acc[t][r] *= inv;
  }

  // =========================================================================
  // Softmax over s of temp1*attn (per q column) — cross-wave via LDS.
  // =========================================================================
  float* parts = (float*)s_u;                 // [14][7][16]
  float* qmax  = (float*)(s_u + 6272);        // [112]
  float* qsum  = (float*)(s_u + 6720);        // [112]

  #pragma unroll
  for (int t = 0; t < NT_W; ++t) {
    float pm = -INFINITY;
    #pragma unroll
    for (int r = 0; r < 8; ++r) {
      int s = s0 + r + 8 * hl;
      if (s < SDIM) pm = fmaxf(pm, temp1 * acc[t][r]);
    }
    pm = fmaxf(pm, __shfl_xor(pm, 16, 32));
    if (lane < 16) parts[(wid * NT_W + t) * 16 + l15] = pm;
  }
  __syncthreads();
  if (tid < WP) {
    int t = tid >> 4, q16 = tid & 15;
    float m = -INFINITY;
    for (int w = 0; w < NWAVES; ++w) m = fmaxf(m, parts[(w * NT_W + t) * 16 + q16]);
    qmax[tid] = m;
  }
  __syncthreads();
  #pragma unroll
  for (int t = 0; t < NT_W; ++t) {
    float qm = qmax[t * 16 + l15];
    float ps = 0.f;
    #pragma unroll
    for (int r = 0; r < 8; ++r) {
      int s = s0 + r + 8 * hl;
      float e = (s < SDIM) ? __expf(temp1 * acc[t][r] - qm) : 0.f;
      acc[t][r] = e;
      ps += e;
    }
    ps += __shfl_xor(ps, 16, 32);
    if (lane < 16) parts[(wid * NT_W + t) * 16 + l15] = ps;
  }
  __syncthreads();
  if (tid < WP) {
    int t = tid >> 4, q16 = tid & 15;
    float s = 0.f;
    for (int w = 0; w < NWAVES; ++w) s += parts[(w * NT_W + t) * 16 + q16];
    qsum[tid] = s;
  }
  __syncthreads();

  // attn2 = e/qsum; emit att_maps (n==b) and bf16 attn2^T tiles for GEMM-2.
  #pragma unroll
  for (int t = 0; t < NT_W; ++t) {
    float inv = 1.f / qsum[t * 16 + l15];
    int q = t * 16 + l15;
    #pragma unroll
    for (int r = 0; r < 8; ++r) {
      int s = s0 + r + 8 * hl;
      float a2 = acc[t][r] * inv;              // 0 for padded s (e was 0)
      if (n == b && q < WDIM && s < SDIM)
        out[1 + (size_t)n * (WDIM * SDIM) + q * SDIM + s] = (q < capn) ? a2 : 0.f;
      if (s < SP2) s_attn2t[q * SP2 + s] = (__bf16)a2;
    }
  }
  __syncthreads();

  // =========================================================================
  // Phase 2: wei[q,c] = attn2^T x img[b], K = s (7 steps of 32, tail padded).
  // Waves pair up per q-tile; c in 12 groups of 64 (img split hi/lo).
  // Cosine-sim partials accumulated against wei tiles in registers.
  // =========================================================================
  __bf16* sbh = (__bf16*)s_u;                 // img frag hi [4][32][16]
  __bf16* sbl = (__bf16*)(s_u + 4096);        // img frag lo
  const int qt2 = wid >> 1;
  const int chalf = wid & 1;

  float pnum[8], pwsq[8], pww[8];
  #pragma unroll
  for (int r = 0; r < 8; ++r) { pnum[r] = 0.f; pwsq[r] = 0.f; pww[r] = 0.f; }

  for (int cg = 0; cg < 12; ++cg) {
    const int cbase0 = cg * 64;
    v8f wacc[2];
    wacc[0] = {}; wacc[1] = {};

    for (int ks = 0; ks < 7; ++ks) {
      const int k0 = ks * 32;
      // ---- cooperative stage of img s-slice into B-fragment layout ----
      #pragma unroll
      for (int i = 0; i < 5; ++i) {
        int e = tid + i * THREADS;
        if (e < 2048) {
          int ct = e >> 9, rem = e & 511, ln = rem >> 4, j = rem & 15;
          int c = cbase0 + ct * 16 + (ln & 15);
          int sr = k0 + ((ln >> 4) << 3) + ((j < 8) ? j : (8 + j));
          float v = (sr < SDIM) ? img[((size_t)b * SDIM + sr) * CDIM + c] : 0.f;
          __bf16 h = (__bf16)v;
          sbh[e] = h;
          sbl[e] = (__bf16)(v - (float)h);
        }
      }
      __syncthreads();

      // ---- A fragment from attn2t (two contiguous 8-half runs) ----
      const int qa = qt2 * 16 + l15;
      v8bf r1 = *(const v8bf*)(s_attn2t + qa * SP2 + k0 + 8 * hl);
      v8bf r2;
      if (ks < 6) {
        r2 = *(const v8bf*)(s_attn2t + qa * SP2 + k0 + 16 + 8 * hl);
      } else {
        #pragma unroll
        for (int j = 0; j < 8; ++j) r2[j] = (__bf16)0.0f;   // s >= 208 pad
      }
      v16bf afrag = __builtin_shufflevector(r1, r2, 0,1,2,3,4,5,6,7,8,9,10,11,12,13,14,15);

      #pragma unroll
      for (int ci = 0; ci < 2; ++ci) {
        int ct = chalf * 2 + ci;
        v16bf bh = *(const v16bf*)(sbh + (ct << 9) + (lane << 4));
        v16bf bl = *(const v16bf*)(sbl + (ct << 9) + (lane << 4));
        wacc[ci] = __builtin_amdgcn_wmma_f32_16x16x32_bf16(false, afrag, false, bh, (short)0, wacc[ci], false, false);
        wacc[ci] = __builtin_amdgcn_wmma_f32_16x16x32_bf16(false, afrag, false, bl, (short)0, wacc[ci], false, false);
      }
      __syncthreads();
    }

    // ---- cosine-sim partials: lane holds one c, 8 q values per tile ----
    #pragma unroll
    for (int ci = 0; ci < 2; ++ci) {
      int c = cbase0 + (chalf * 2 + ci) * 16 + l15;
      #pragma unroll
      for (int r = 0; r < 8; ++r) {
        int q = qt2 * 16 + 8 * hl + r;
        float wv = (q < WDIM) ? words[((size_t)n * CDIM + c) * WDIM + q] : 0.f;
        float we = wacc[ci][r];
        pnum[r] += wv * we;
        pwsq[r] += wv * wv;
        pww[r]  += we * we;
      }
    }
  }

  // reduce over the 16 c-lanes, then combine wave pairs via LDS atomics
  #pragma unroll
  for (int r = 0; r < 8; ++r) {
    #pragma unroll
    for (int off = 1; off < 16; off <<= 1) {
      pnum[r] += __shfl_xor(pnum[r], off, 16);
      pwsq[r] += __shfl_xor(pwsq[r], off, 16);
      pww[r]  += __shfl_xor(pww[r],  off, 16);
    }
  }
  if (l15 == 0) {
    #pragma unroll
    for (int r = 0; r < 8; ++r) {
      int q = qt2 * 16 + 8 * hl + r;
      atomicAdd(&s_red[q * 3 + 0], pnum[r]);
      atomicAdd(&s_red[q * 3 + 1], pwsq[r]);
      atomicAdd(&s_red[q * 3 + 2], pww[r]);
    }
  }
  __syncthreads();

  if (tid < WP) {
    int q = tid;
    float d = sqrtf(s_red[q * 3 + 1]) * sqrtf(s_red[q * 3 + 2]);
    d = fmaxf(d, 1e-8f);
    float sim = s_red[q * 3 + 0] / d;
    float contrib = (q < capn) ? __expf(temp2 * sim) : 0.f;
    atomicAdd(&s_simsum, contrib);
  }
  __syncthreads();
  if (tid == 0) simlog[(size_t)n * NB + b] = logf(s_simsum);
}

// ---------------------------------------------------------------------------
// Loss kernel: similarities[b][n] = temp3*simlog[n,b]; symmetric LSE loss.
// ---------------------------------------------------------------------------
__global__ void damsm_loss_kernel(const float* __restrict__ simlog,
                                  const float* __restrict__ t3p,
                                  float* __restrict__ out)
{
  const int lane = threadIdx.x;   // 0..31
  const float t3 = t3p[0];

  // row (b = lane) over n
  float rmax = -INFINITY;
  for (int n2 = 0; n2 < NB; ++n2) rmax = fmaxf(rmax, t3 * simlog[n2 * NB + lane]);
  float rsum = 0.f;
  for (int n2 = 0; n2 < NB; ++n2) rsum += __expf(t3 * simlog[n2 * NB + lane] - rmax);
  const float diag = t3 * simlog[lane * NB + lane];
  float p0 = logf(rsum) + rmax - diag;

  // column (n = lane) over b
  float cmax = -INFINITY;
  for (int b2 = 0; b2 < NB; ++b2) cmax = fmaxf(cmax, t3 * simlog[lane * NB + b2]);
  float csum = 0.f;
  for (int b2 = 0; b2 < NB; ++b2) csum += __expf(t3 * simlog[lane * NB + b2] - cmax);
  float p1 = logf(csum) + cmax - diag;

  float v = p0 + p1;
  for (int off = 1; off < 32; off <<= 1) v += __shfl_xor(v, off, 32);
  if (lane == 0) out[0] = v * (1.0f / 64.0f);  // ((sum p0)/32 + (sum p1)/32) / 2
}

extern "C" void kernel_launch(void* const* d_in, const int* in_sizes, int n_in,
                              void* d_out, int out_size, void* d_ws, size_t ws_size,
                              hipStream_t stream) {
  (void)in_sizes; (void)n_in; (void)out_size; (void)ws_size;
  const float* img   = (const float*)d_in[0];
  const float* words = (const float*)d_in[1];
  const int*   caps  = (const int*)d_in[2];
  const float* t1    = (const float*)d_in[3];
  const float* t2    = (const float*)d_in[4];
  const float* t3    = (const float*)d_in[5];
  float* out = (float*)d_out;
  float* simlog = (float*)d_ws;     // NB*NB floats

  dim3 grid(NB, NB);
  damsm_attn_kernel<<<grid, THREADS, 0, stream>>>(img, words, caps, t1, t2, out, simlog);
  damsm_loss_kernel<<<1, 32, 0, stream>>>(simlog, t3, out);
}
